// PureLearningLocalCostVolume_66606352827286
// MI455X (gfx1250) — matrix operand order repaired and checked by reading, not compiled, not run on hardware
//
#include <hip/hip_runtime.h>
#include <hip/hip_bf16.h>

typedef __attribute__((ext_vector_type(16))) _Float16 v16h;
typedef __attribute__((ext_vector_type(8)))  _Float16 v8h;
typedef __attribute__((ext_vector_type(8)))  float    v8f;

#define HH 320
#define WW 640
#define DD 24

// ---------------------------------------------------------------------------
// gfx1250 async global->LDS copy (16B per lane), tracked by ASYNCcnt.
// ---------------------------------------------------------------------------
__device__ __forceinline__ void async_copy_b128(void* lds, const void* gaddr) {
  unsigned l = (unsigned)(uintptr_t)lds;               // low 32 bits = LDS offset
  unsigned long long g = (unsigned long long)(uintptr_t)gaddr;
  asm volatile("global_load_async_to_lds_b128 %0, %1, off" :: "v"(l), "v"(g) : "memory");
}
__device__ __forceinline__ void async_wait0() {
  asm volatile("s_wait_asynccnt 0x0" ::: "memory");
}

__device__ __forceinline__ v16h cat16(v8h a, v8h b) {
  return __builtin_shufflevector(a, b, 0, 1, 2, 3, 4, 5, 6, 7, 8, 9, 10, 11, 12, 13, 14, 15);
}

// ---------------------------------------------------------------------------
// Weight pack: fp32 OIHW -> fp16 [Cout][KPad]; optional tap-major reorder
// (k = tap*Cin + cin) so im2col B-fragments are cin-contiguous.
// ---------------------------------------------------------------------------
__global__ void pack_w_kernel(const float* __restrict__ src, _Float16* __restrict__ dst,
                              int Cout, int K, int KPad, int Cin, int tap_major) {
  int i = blockIdx.x * 256 + threadIdx.x;
  if (i >= Cout * KPad) return;
  int o = i / KPad, k = i - o * KPad;
  _Float16 val = (_Float16)0.f;
  if (k < K) {
    int srck = k;
    if (tap_major) { int tap = k / Cin, c = k - tap * Cin; srck = c * 9 + tap; }
    val = (_Float16)src[o * K + srck];
  }
  dst[i] = val;
}

// ---------------------------------------------------------------------------
// Prep: error map (bilinear disparity warp) + uvol = softmax(cv)*d*(d-disp)^2
// ---------------------------------------------------------------------------
__global__ void prep_kernel(const float* __restrict__ cv,    // [D][H][W]
                            const float* __restrict__ disp,  // [H][W]
                            const float* __restrict__ left,  // [3][H][W]
                            const float* __restrict__ right, // [3][H][W]
                            _Float16* __restrict__ err,      // [3][H][W]
                            _Float16* __restrict__ uvol) {   // [D][H][W]
  const int HW = HH * WW;
  int idx = blockIdx.x * blockDim.x + threadIdx.x;
  if (idx >= HW) return;
  int x = idx % WW;
  int rowoff = idx - x;
  float dsp = disp[idx];
  float xc  = (float)x - dsp;
  float x0f = floorf(xc);
  float w1  = xc - x0f;
  int x0i = min(max((int)x0f, 0), WW - 1);
  int x1i = min(max((int)x0f + 1, 0), WW - 1);
  float valid = (xc >= 0.f && xc <= (float)(WW - 1)) ? 1.f : 0.f;
#pragma unroll
  for (int c = 0; c < 3; ++c) {
    float g0 = right[c * HW + rowoff + x0i];
    float g1 = right[c * HW + rowoff + x1i];
    float wr = (g0 * (1.f - w1) + g1 * w1) * valid;
    err[c * HW + idx] = (_Float16)(wr - left[c * HW + idx]);
  }
  float vals[DD];
  float mx = -1e30f;
#pragma unroll
  for (int d = 0; d < DD; ++d) { vals[d] = cv[d * HW + idx]; mx = fmaxf(mx, vals[d]); }
  float ssum = 0.f;
#pragma unroll
  for (int d = 0; d < DD; ++d) { vals[d] = __expf(vals[d] - mx); ssum += vals[d]; }
  float inv = 1.f / ssum;
#pragma unroll
  for (int d = 0; d < DD; ++d) {
    float dd = (float)d, df = dd - dsp;
    uvol[d * HW + idx] = (_Float16)(vals[d] * inv * dd * df * df);
  }
}

// ---------------------------------------------------------------------------
// Small-Cin conv (Cin=3/24): scalar im2col gathers, cin-major K (zero padded).
// ---------------------------------------------------------------------------
template <int CIN, int COUT, int KPAD>
__global__ __launch_bounds__(128) void conv3x3_wmma(
    const _Float16* __restrict__ in, const _Float16* __restrict__ wbuf,
    const float* __restrict__ g, const float* __restrict__ b,
    const float* __restrict__ m, const float* __restrict__ v,
    _Float16* __restrict__ out) {
  __shared__ _Float16 s_w[COUT * KPAD];
  __shared__ _Float16 s_in[CIN * 3 * 66];
  __shared__ float    s_bn[COUT * 2];
  const int HW = HH * WW;
  int tid = threadIdx.x;
  int x0  = blockIdx.x * 64;
  int y   = blockIdx.y;

  for (int i = tid; i < COUT * KPAD / 8; i += 128)
    async_copy_b128(&s_w[i * 8], wbuf + i * 8);
  for (int i = tid; i < CIN * 3 * 66; i += 128) {
    int c = i / 198, rem = i - c * 198;
    int ry = rem / 66, rx = rem - ry * 66;
    int yy = y + ry - 1, xx = x0 + rx - 1;
    s_in[i] = (yy >= 0 && yy < HH && xx >= 0 && xx < WW)
                  ? in[c * HW + yy * WW + xx] : (_Float16)0.f;
  }
  if (tid < COUT) {
    float sc = g[tid] * rsqrtf(v[tid] + 1e-5f);
    s_bn[2 * tid]     = sc;
    s_bn[2 * tid + 1] = b[tid] - m[tid] * sc;
  }
  async_wait0();
  __syncthreads();

  int lane = tid & 31;
  int lh = lane >> 4, lp = lane & 15;
  int px = (tid >> 5) * 16 + lp;

  for (int ct = 0; ct < COUT / 16; ++ct) {
    v8f acc = {0.f, 0.f, 0.f, 0.f, 0.f, 0.f, 0.f, 0.f};
    int crow = (ct * 16 + lp) * KPAD;
    for (int k0 = 0; k0 < KPAD; k0 += 32) {
      const _Float16* wp = &s_w[crow + k0 + (lh << 4)];
      v16h af = cat16(*(const v8h*)wp, *(const v8h*)(wp + 8));
      v16h bf;
#pragma unroll
      for (int e = 0; e < 16; ++e) {
        int kg = k0 + (lh << 4) + e;
        _Float16 val = (_Float16)0.f;
        if (kg < CIN * 9) {
          int cin = kg / 9, r9 = kg - cin * 9;
          int dy = r9 / 3, dx = r9 - dy * 3;
          val = s_in[(cin * 3 + dy) * 66 + px + dx];
        }
        bf[e] = val;
      }
      acc = __builtin_amdgcn_wmma_f32_16x16x32_f16(false, af, false, bf,
                                                   (short)0, acc, false, false);
    }
    int xo = x0 + px;
#pragma unroll
    for (int r = 0; r < 8; ++r) {
      int c = ct * 16 + lh * 8 + r;
      float o = acc[r] * s_bn[2 * c] + s_bn[2 * c + 1];
      out[c * HW + y * WW + xo] = (_Float16)fmaxf(o, 0.f);
    }
  }
}

// ---------------------------------------------------------------------------
// Fast conv (Cin multiple of 16): tap-major weights + channel-last LDS patch
// -> both WMMA fragments are two contiguous 16B LDS loads.
// TILEW: stage weights 16 output rows at a time (keeps Cin=64 under LDS cap).
// ---------------------------------------------------------------------------
template <int CIN, int COUT, bool TILEW>
__global__ __launch_bounds__(128) void conv3x3_wmma_fast(
    const _Float16* __restrict__ in, const _Float16* __restrict__ wbuf,
    const float* __restrict__ g, const float* __restrict__ b,
    const float* __restrict__ m, const float* __restrict__ v,
    _Float16* __restrict__ out) {
  constexpr int K     = CIN * 9;
  constexpr int CINP  = CIN + 8;        // padded row: 16B-aligned, bank-rotated
  constexpr int WROWS = TILEW ? 16 : COUT;
  __shared__ _Float16 s_w[WROWS * K];
  __shared__ _Float16 s_in[198 * CINP];
  __shared__ float    s_bn[COUT * 2];
  const int HW = HH * WW;
  int tid = threadIdx.x;
  int x0  = blockIdx.x * 64;
  int y   = blockIdx.y;

  if (!TILEW)
    for (int i = tid; i < COUT * K / 8; i += 128)
      async_copy_b128(&s_w[i * 8], wbuf + i * 8);
  for (int i = tid; i < CIN * 198; i += 128) {
    int c = i / 198, rem = i - c * 198;
    int ry = rem / 66, rx = rem - ry * 66;
    int yy = y + ry - 1, xx = x0 + rx - 1;
    _Float16 val = (yy >= 0 && yy < HH && xx >= 0 && xx < WW)
                       ? in[c * HW + yy * WW + xx] : (_Float16)0.f;
    s_in[rem * CINP + c] = val;
  }
  if (tid < COUT) {
    float sc = g[tid] * rsqrtf(v[tid] + 1e-5f);
    s_bn[2 * tid]     = sc;
    s_bn[2 * tid + 1] = b[tid] - m[tid] * sc;
  }
  async_wait0();
  __syncthreads();

  int lane = tid & 31;
  int lh = lane >> 4, lp = lane & 15;
  int px = (tid >> 5) * 16 + lp;

  for (int ct = 0; ct < COUT / 16; ++ct) {
    if (TILEW) {
      __syncthreads();  // previous tile fully consumed
      for (int i = tid; i < 16 * K / 8; i += 128)
        async_copy_b128(&s_w[i * 8], wbuf + ct * 16 * K + i * 8);
      async_wait0();
      __syncthreads();
    }
    v8f acc = {0.f, 0.f, 0.f, 0.f, 0.f, 0.f, 0.f, 0.f};
    int crow = (TILEW ? lp : (ct * 16 + lp)) * K;
#pragma unroll
    for (int k0 = 0; k0 < K; k0 += 32) {
      const _Float16* wp = &s_w[crow + k0 + (lh << 4)];
      v16h af = cat16(*(const v8h*)wp, *(const v8h*)(wp + 8));
      int run  = k0 + (lh << 4);
      int tap  = run / CIN;
      int cinb = run - tap * CIN;
      int dy = tap / 3, dx = tap - dy * 3;
      const _Float16* bp = &s_in[(dy * 66 + px + dx) * CINP + cinb];
      v16h bf = cat16(*(const v8h*)bp, *(const v8h*)(bp + 8));
      acc = __builtin_amdgcn_wmma_f32_16x16x32_f16(false, af, false, bf,
                                                   (short)0, acc, false, false);
    }
    int xo = x0 + px;
#pragma unroll
    for (int r = 0; r < 8; ++r) {
      int c = ct * 16 + lh * 8 + r;
      float o = acc[r] * s_bn[2 * c] + s_bn[2 * c + 1];
      out[c * HW + y * WW + xo] = (_Float16)fmaxf(o, 0.f);
    }
  }
}

// ---------------------------------------------------------------------------
// Fused ResBlock tail: out = relu( bn2(conv3x3(h, rb2)) + bnd(conv1x1(cat, rbd)) )
// rb2 uses fast (tap-major/channel-last) path; 1x1 B-fragments are contiguous.
// ---------------------------------------------------------------------------
__global__ __launch_bounds__(128) void resblock_tail_wmma(
    const _Float16* __restrict__ hin,  // [32][H][W]
    const _Float16* __restrict__ cat,  // [64][H][W]
    const _Float16* __restrict__ w2,   // [32][288] tap-major
    const _Float16* __restrict__ wd,   // [32][64]
    const float* g2, const float* b2, const float* m2, const float* v2,
    const float* gd, const float* bd, const float* md, const float* vd,
    _Float16* __restrict__ out) {      // [32][H][W]
  __shared__ _Float16 s_w2[32 * 288];
  __shared__ _Float16 s_wd[32 * 64];
  __shared__ _Float16 s_h[198 * 40];   // channel-last, Cin=32 padded to 40
  __shared__ _Float16 s_c[64 * 72];    // [px][cin], 64 padded to 72
  __shared__ float    s_bn2[64], s_bnd[64];
  const int HW = HH * WW;
  int tid = threadIdx.x;
  int x0  = blockIdx.x * 64;
  int y   = blockIdx.y;

  for (int i = tid; i < 32 * 288 / 8; i += 128) async_copy_b128(&s_w2[i * 8], w2 + i * 8);
  for (int i = tid; i < 32 * 64 / 8; i += 128)  async_copy_b128(&s_wd[i * 8], wd + i * 8);
  for (int i = tid; i < 32 * 198; i += 128) {
    int c = i / 198, rem = i - c * 198;
    int ry = rem / 66, rx = rem - ry * 66;
    int yy = y + ry - 1, xx = x0 + rx - 1;
    _Float16 val = (yy >= 0 && yy < HH && xx >= 0 && xx < WW)
                       ? hin[c * HW + yy * WW + xx] : (_Float16)0.f;
    s_h[rem * 40 + c] = val;
  }
  for (int i = tid; i < 64 * 64; i += 128) {  // 1x1: center row only
    int c = i >> 6, rx = i & 63;
    s_c[rx * 72 + c] = cat[c * HW + y * WW + x0 + rx];
  }
  if (tid < 32) {
    float sc = g2[tid] * rsqrtf(v2[tid] + 1e-5f);
    s_bn2[2 * tid] = sc; s_bn2[2 * tid + 1] = b2[tid] - m2[tid] * sc;
    float sd = gd[tid] * rsqrtf(vd[tid] + 1e-5f);
    s_bnd[2 * tid] = sd; s_bnd[2 * tid + 1] = bd[tid] - md[tid] * sd;
  }
  async_wait0();
  __syncthreads();

  int lane = tid & 31;
  int lh = lane >> 4, lp = lane & 15;
  int px = (tid >> 5) * 16 + lp;

  for (int ct = 0; ct < 2; ++ct) {
    v8f c1 = {0.f, 0.f, 0.f, 0.f, 0.f, 0.f, 0.f, 0.f};
    v8f c2 = {0.f, 0.f, 0.f, 0.f, 0.f, 0.f, 0.f, 0.f};
    int crow2 = (ct * 16 + lp) * 288;
#pragma unroll
    for (int k0 = 0; k0 < 288; k0 += 32) {   // 3x3 over h, Cin=32 tap-major
      const _Float16* wp = &s_w2[crow2 + k0 + (lh << 4)];
      v16h af = cat16(*(const v8h*)wp, *(const v8h*)(wp + 8));
      int run  = k0 + (lh << 4);
      int tap  = run >> 5;
      int cinb = run & 31;
      int dy = tap / 3, dx = tap - dy * 3;
      const _Float16* bp = &s_h[(dy * 66 + px + dx) * 40 + cinb];
      v16h bf = cat16(*(const v8h*)bp, *(const v8h*)(bp + 8));
      c1 = __builtin_amdgcn_wmma_f32_16x16x32_f16(false, af, false, bf,
                                                  (short)0, c1, false, false);
    }
    int crowd = (ct * 16 + lp) * 64;
#pragma unroll
    for (int k0 = 0; k0 < 64; k0 += 32) {    // 1x1 over cat, Cin=64
      const _Float16* wp = &s_wd[crowd + k0 + (lh << 4)];
      v16h af = cat16(*(const v8h*)wp, *(const v8h*)(wp + 8));
      const _Float16* bp = &s_c[px * 72 + k0 + (lh << 4)];
      v16h bf = cat16(*(const v8h*)bp, *(const v8h*)(bp + 8));
      c2 = __builtin_amdgcn_wmma_f32_16x16x32_f16(false, af, false, bf,
                                                  (short)0, c2, false, false);
    }
    int xo = x0 + px;
#pragma unroll
    for (int r = 0; r < 8; ++r) {
      int c = ct * 16 + lh * 8 + r;
      float o = c1[r] * s_bn2[2 * c] + s_bn2[2 * c + 1]
              + c2[r] * s_bnd[2 * c] + s_bnd[2 * c + 1];
      out[c * HW + y * WW + xo] = (_Float16)fmaxf(o, 0.f);
    }
  }
}

// ---------------------------------------------------------------------------
// Final: off = relu(conv3x3(f, off_w)+off_b) (2 ch, VALU),
// then 11-point local cost volume gather/lerp + softmax + expectation.
// ---------------------------------------------------------------------------
__global__ __launch_bounds__(128) void final_kernel(
    const _Float16* __restrict__ f,    // [16][H][W]
    const float* __restrict__ offw,    // [2][16][3][3]
    const float* __restrict__ offb,    // [2]
    const float* __restrict__ cv,      // [D][H][W]
    float* __restrict__ out) {         // [H][W]
  __shared__ _Float16 s_f[16 * 3 * 130];
  __shared__ float    s_w[288];
  __shared__ float    s_b[2];
  const int HW = HH * WW;
  int tid = threadIdx.x;
  int x0  = blockIdx.x * 128;
  int y   = blockIdx.y;
  for (int i = tid; i < 72; i += 128)
    async_copy_b128(&s_w[i * 4], offw + i * 4);
  if (tid < 2) s_b[tid] = offb[tid];
  for (int i = tid; i < 16 * 3 * 130; i += 128) {
    int c = i / 390, rem = i - c * 390;
    int ry = rem / 130, rx = rem - ry * 130;
    int yy = y + ry - 1, xx = x0 + rx - 1;
    s_f[i] = (yy >= 0 && yy < HH && xx >= 0 && xx < WW)
                 ? f[c * HW + yy * WW + xx] : (_Float16)0.f;
  }
  async_wait0();
  __syncthreads();
  int x = x0 + tid;
  float a0 = s_b[0], a1 = s_b[1];
#pragma unroll
  for (int c = 0; c < 16; ++c)
#pragma unroll
    for (int dy = 0; dy < 3; ++dy)
#pragma unroll
      for (int dx = 0; dx < 3; ++dx) {
        float vv = (float)s_f[(c * 3 + dy) * 130 + tid + dx];
        int k = c * 9 + dy * 3 + dx;
        a0 += vv * s_w[k];
        a1 += vv * s_w[144 + k];
      }
  float lower = fmaxf(a0, 0.f);
  float upper = fminf(fmaxf(a1, 0.f), (float)DD);
  float interval = (upper - lower) * (1.f / 10.f);
  float sv[11], lv[11];
  float mx = -1e30f;
#pragma unroll
  for (int k = 0; k <= 10; ++k) {
    float s  = lower + (float)k * interval;
    float cf = fminf(fmaxf(ceilf(s), 0.f), (float)(DD - 1));
    float ff = fminf(fmaxf(floorf(s), 0.f), (float)(DD - 1));
    int ci = (int)cf, fi = (int)ff;
    float fr = cf - s;
    float cr = 1.f - fr;
    float fv  = cv[fi * HW + y * WW + x];
    float cvv = cv[ci * HW + y * WW + x];
    float l = fv * fr + cvv * cr;
    sv[k] = s; lv[k] = l;
    mx = fmaxf(mx, l);
  }
  float ssum = 0.f, acc = 0.f;
#pragma unroll
  for (int k = 0; k <= 10; ++k) {
    float e = __expf(lv[k] - mx);
    ssum += e;
    acc  += e * sv[k];
  }
  out[y * WW + x] = acc / ssum;
}

// ---------------------------------------------------------------------------
// Host side
// ---------------------------------------------------------------------------
static int find_size(const int* s, int n, int sz, int occ) {
  int seen = 0;
  for (int i = 0; i < n; ++i)
    if (s[i] == sz) { if (seen == occ) return i; ++seen; }
  return -1;
}

extern "C" void kernel_launch(void* const* d_in, const int* in_sizes, int n_in,
                              void* d_out, int out_size, void* d_ws, size_t ws_size,
                              hipStream_t stream) {
  const size_t HW = (size_t)HH * WW;
  int i_cv   = find_size(in_sizes, n_in, 8 * DD * (int)HW, 0);
  int i_disp = find_size(in_sizes, n_in, 8 * (int)HW, 0);
  int i_left = find_size(in_sizes, n_in, 8 * 3 * (int)HW, 0);
  int i_right= find_size(in_sizes, n_in, 8 * 3 * (int)HW, 1);
  int i_ue   = find_size(in_sizes, n_in, 32 * 24 * 9, 0);
  int i_de   = find_size(in_sizes, n_in, 32 * 3 * 9, 0);
  int i_rb1  = find_size(in_sizes, n_in, 32 * 64 * 9, 0);
  int i_rb2  = find_size(in_sizes, n_in, 32 * 32 * 9, 0);
  int i_rbd  = find_size(in_sizes, n_in, 32 * 64, 0);
  int i_db   = find_size(in_sizes, n_in, 16 * 32 * 9, 0);
  int i_offw = find_size(in_sizes, n_in, 2 * 16 * 9, 0);
  int i_offb = find_size(in_sizes, n_in, 2, 0);
  bool bn_after = (i_de + 1 < n_in) && (in_sizes[i_de + 1] == 32);
  const float *G[6], *Bb[6], *M[6], *V[6];
  int widx[6] = {i_de, i_ue, i_rb1, i_rb2, i_rbd, i_db};
  for (int j = 0; j < 6; ++j) {
    int wi = widx[j];
    if (bn_after) {
      G[j] = (const float*)d_in[wi + 1]; Bb[j] = (const float*)d_in[wi + 2];
      M[j] = (const float*)d_in[wi + 3]; V[j]  = (const float*)d_in[wi + 4];
    } else {
      Bb[j] = (const float*)d_in[wi - 4]; G[j] = (const float*)d_in[wi - 3];
      M[j] = (const float*)d_in[wi - 2];  V[j] = (const float*)d_in[wi - 1];
    }
  }
  const float* cv    = (const float*)d_in[i_cv];
  const float* disp  = (const float*)d_in[i_disp];
  const float* left  = (const float*)d_in[i_left];
  const float* right = (const float*)d_in[i_right];

  char* base = (char*)d_ws;
  size_t off = 0;
  auto alloc = [&](size_t bytes) -> char* {
    char* p = base + off;
    off = (off + bytes + 255) & ~(size_t)255;
    return p;
  };
  _Float16* w_de  = (_Float16*)alloc(32 * 32 * 2);
  _Float16* w_ue  = (_Float16*)alloc(32 * 224 * 2);
  _Float16* w_rb1 = (_Float16*)alloc(32 * 576 * 2);
  _Float16* w_rb2 = (_Float16*)alloc(32 * 288 * 2);
  _Float16* w_rbd = (_Float16*)alloc(32 * 64 * 2);
  _Float16* w_db  = (_Float16*)alloc(16 * 288 * 2);
  _Float16* errb  = (_Float16*)alloc(3 * HW * 2);
  _Float16* uvol  = (_Float16*)alloc(24 * HW * 2);
  _Float16* catb  = (_Float16*)alloc(64 * HW * 2);
  _Float16* hbuf  = (_Float16*)alloc(32 * HW * 2);
  _Float16* resb  = (_Float16*)alloc(32 * HW * 2);
  _Float16* fbuf  = (_Float16*)alloc(16 * HW * 2);
  (void)ws_size;

  auto pack = [&](const float* src, _Float16* dst, int Cout, int K, int KPad,
                  int Cin, int tapm) {
    int tot = Cout * KPad;
    pack_w_kernel<<<(tot + 255) / 256, 256, 0, stream>>>(src, dst, Cout, K, KPad, Cin, tapm);
  };
  pack((const float*)d_in[i_de],  w_de,  32, 27,  32,  3,  0);
  pack((const float*)d_in[i_ue],  w_ue,  32, 216, 224, 24, 0);
  pack((const float*)d_in[i_rb1], w_rb1, 32, 576, 576, 64, 1);
  pack((const float*)d_in[i_rb2], w_rb2, 32, 288, 288, 32, 1);
  pack((const float*)d_in[i_rbd], w_rbd, 32, 64,  64,  64, 0);
  pack((const float*)d_in[i_db],  w_db,  16, 288, 288, 32, 1);

  dim3 cgrid(WW / 64, HH);
  dim3 fgrid(WW / 128, HH);
  for (int b = 0; b < 8; ++b) {
    const float* cv_b    = cv    + (size_t)b * DD * HW;
    const float* disp_b  = disp  + (size_t)b * HW;
    const float* left_b  = left  + (size_t)b * 3 * HW;
    const float* right_b = right + (size_t)b * 3 * HW;
    float*       out_b   = (float*)d_out + (size_t)b * HW;

    prep_kernel<<<(int)((HW + 255) / 256), 256, 0, stream>>>(
        cv_b, disp_b, left_b, right_b, errb, uvol);
    conv3x3_wmma<3, 32, 32><<<cgrid, 128, 0, stream>>>(
        errb, w_de, G[0], Bb[0], M[0], V[0], catb);
    conv3x3_wmma<24, 32, 224><<<cgrid, 128, 0, stream>>>(
        uvol, w_ue, G[1], Bb[1], M[1], V[1], catb + 32 * HW);
    conv3x3_wmma_fast<64, 32, true><<<cgrid, 128, 0, stream>>>(
        catb, w_rb1, G[2], Bb[2], M[2], V[2], hbuf);
    resblock_tail_wmma<<<cgrid, 128, 0, stream>>>(
        hbuf, catb, w_rb2, w_rbd,
        G[3], Bb[3], M[3], V[3], G[4], Bb[4], M[4], V[4], resb);
    conv3x3_wmma_fast<32, 16, false><<<cgrid, 128, 0, stream>>>(
        resb, w_db, G[5], Bb[5], M[5], V[5], fbuf);
    final_kernel<<<fgrid, 128, 0, stream>>>(
        fbuf, (const float*)d_in[i_offw], (const float*)d_in[i_offb], cv_b, out_b);
  }
}
